// Cheb_GCNN_3p_81063212744711
// MI455X (gfx1250) — matrix operand order repaired and checked by reading, not compiled, not run on hardware
//
#include <hip/hip_runtime.h>

#define FEAT 64
#define OUTF 2
#define LMAX 2.0f
#define BN_EPS 1e-5f

typedef __attribute__((ext_vector_type(16))) _Float16 v16h;
typedef __attribute__((ext_vector_type(8)))  _Float16 v8h;
typedef __attribute__((ext_vector_type(8)))  float    v8f;

// ---------------------------------------------------------------------------
// Dense GEMM:  C[NR,64] (+)= A16[NR,64] @ Wt^T   (f16 WMMA, f32 accumulate)
//   A16 : row-major f16 activations (NR rows, NR % 32 == 0, padded workspace)
//   Bt  : f16 weights pre-transposed: Bt[n][k] = W[k][n]  (64x64)
// Block = 256 threads = 8 waves; block covers 32 rows x 64 cols, one 16x16
// tile per wave, K=64 -> two v_wmma_f32_16x16x32_f16.  No guards: all row
// tiles (incl. padding rows) are backed by allocated workspace.
// VGPR layouts used here (wave32; contiguous k, not the ISA table's k order):
//   A 16x32 f16 : lane half h, elems[0..15]=K h*16..h*16+15 (row m = lane&15)
//   B 32x16 f16 : lane half h, elems[0..15]=K h*16..h*16+15 (col n = lane&15)
//   C/D 16x16   : vgpr r -> row m0 + h*8 + r, col n0 + (lane&15)
// ---------------------------------------------------------------------------
__global__ __launch_bounds__(256)
void gemm_n64_f16wmma(const _Float16* __restrict__ A16,
                      const _Float16* __restrict__ Bt,
                      const float* __restrict__ bias,
                      float* __restrict__ C,
                      int initFromBias)
{
    const int wave = threadIdx.x >> 5;
    const int lane = threadIdx.x & 31;
    const int half = lane >> 4;
    const int l16  = lane & 15;
    const int m0 = blockIdx.x * 32 + (wave & 1) * 16;
    const int n0 = (wave >> 1) * 16;
    const int n  = n0 + l16;

    const _Float16* arow = A16 + (size_t)(m0 + l16) * FEAT;
    const _Float16* brow = Bt  + (size_t)n * FEAT;

    // A fragments: two contiguous 16B chunks per K-block
    const v8h a0lo = *(const v8h*)(arow + half * 16);
    const v8h a0hi = *(const v8h*)(arow + 8 + half * 16);
    const v8h a1lo = *(const v8h*)(arow + 32 + half * 16);
    const v8h a1hi = *(const v8h*)(arow + 40 + half * 16);
    // B fragments: 16 contiguous f16 per K-block (thanks to transposed Wt)
    const v16h b0 = *(const v16h*)(brow + half * 16);
    const v16h b1 = *(const v16h*)(brow + 32 + half * 16);

    v16h a0, a1;
#pragma unroll
    for (int i = 0; i < 8; ++i) {
        a0[i] = a0lo[i]; a0[8 + i] = a0hi[i];
        a1[i] = a1lo[i]; a1[8 + i] = a1hi[i];
    }

    v8f c;
    if (initFromBias) {
        const float bv = bias[n];
#pragma unroll
        for (int r = 0; r < 8; ++r) c[r] = bv;
    } else {
#pragma unroll
        for (int r = 0; r < 8; ++r)
            c[r] = C[(size_t)(m0 + half * 8 + r) * FEAT + n];
    }

    c = __builtin_amdgcn_wmma_f32_16x16x32_f16(false, a0, false, b0,
                                               (short)0, c, false, false);
    c = __builtin_amdgcn_wmma_f32_16x16x32_f16(false, a1, false, b1,
                                               (short)0, c, false, false);

#pragma unroll
    for (int r = 0; r < 8; ++r)
        C[(size_t)(m0 + half * 8 + r) * FEAT + n] = c[r];
}

// One-time weight prep: Wt[k][n][kd] = (f16) W[k][kd][n]   (kmats 64x64 mats)
__global__ void wprep_kernel(const float* __restrict__ W,
                             _Float16* __restrict__ Wt, int total)
{
    int i = blockIdx.x * blockDim.x + threadIdx.x;
    if (i >= total) return;
    int k  = i >> 12;
    int nn = (i >> 6) & 63;
    int kd = i & 63;
    Wt[i] = (_Float16)W[((size_t)k << 12) + (size_t)kd * FEAT + nn];
}

// f32 -> f16 shadow copy
__global__ void cvt_f16_kernel(const float* __restrict__ src,
                               _Float16* __restrict__ dst, size_t count)
{
    size_t i = (size_t)blockIdx.x * blockDim.x + threadIdx.x;
    if (i < count) dst[i] = (_Float16)src[i];
}

// ------------------------- graph normalization ----------------------------
__global__ void deg_kernel(const int* __restrict__ row,
                           const float* __restrict__ ew,
                           float* __restrict__ deg, int E)
{
    int e = blockIdx.x * blockDim.x + threadIdx.x;
    if (e < E) atomicAdd(&deg[row[e]], ew[e]);
}

__global__ void dis_kernel(const float* __restrict__ deg,
                           float* __restrict__ dis, int n)
{
    int i = blockIdx.x * blockDim.x + threadIdx.x;
    if (i < n) {
        float d = deg[i];
        dis[i] = (d > 0.0f) ? rsqrtf(d) : 0.0f;
    }
}

__global__ void wnorm_kernel(const int* __restrict__ row,
                             const int* __restrict__ col,
                             const float* __restrict__ ew,
                             const float* __restrict__ dis,
                             float* __restrict__ wn, int E)
{
    int e = blockIdx.x * blockDim.x + threadIdx.x;
    if (e < E)
        wn[e] = -(dis[row[e]] * ew[e] * dis[col[e]]) * (2.0f / LMAX);
}

// ------------------- sparse L_hat @ H (scatter, atomics) -------------------
__global__ void scatter_lmul(const int* __restrict__ row,
                             const int* __restrict__ col,
                             const float* __restrict__ wn,
                             const float* __restrict__ h,
                             float* __restrict__ s, int E)
{
    long long tid = (long long)blockIdx.x * blockDim.x + threadIdx.x;
    long long e = tid >> 4;
    if (e >= (long long)E) return;
    int c = (int)(tid & 15) * 4;
    int r  = row[e];
    int cl = col[e];
    float we = wn[e];
    const float4 hv = *(const float4*)(h + (size_t)r * FEAT + c);
    float* sp = s + (size_t)cl * FEAT + c;
    atomicAdd(sp + 0, we * hv.x);
    atomicAdd(sp + 1, we * hv.y);
    atomicAdd(sp + 2, we * hv.z);
    atomicAdd(sp + 3, we * hv.w);
}

// txn = alpha*(s + diag*txb) + beta*txa ; also emits f16 copy for the GEMM
__global__ void cheb_combine(const float* __restrict__ s,
                             const float* __restrict__ txb,
                             const float* __restrict__ txa,
                             float* __restrict__ txn,
                             _Float16* __restrict__ txn16,
                             float alpha, float beta, float diag, size_t count)
{
    size_t i = (size_t)blockIdx.x * blockDim.x + threadIdx.x;
    if (i < count) {
        float v = alpha * (s[i] + diag * txb[i]) + beta * txa[i];
        txn[i] = v;
        txn16[i] = (_Float16)v;
    }
}

// h = BN(relu(acc)) (eval mode); also emits f16 copy for the next GEMM
__global__ void relu_bn_kernel(const float* __restrict__ acc,
                               const float* __restrict__ g,
                               const float* __restrict__ b,
                               const float* __restrict__ mean,
                               const float* __restrict__ var,
                               float* __restrict__ h,
                               _Float16* __restrict__ h16, size_t count)
{
    size_t i = (size_t)blockIdx.x * blockDim.x + threadIdx.x;
    if (i < count) {
        int f = (int)(i & (FEAT - 1));
        float v = fmaxf(acc[i], 0.0f);
        float o = g[f] * (v - mean[f]) * rsqrtf(var[f] + BN_EPS) + b[f];
        h[i] = o;
        h16[i] = (_Float16)o;
    }
}

__global__ void relu_inplace(float* __restrict__ acc, size_t count)
{
    size_t i = (size_t)blockIdx.x * blockDim.x + threadIdx.x;
    if (i < count) acc[i] = fmaxf(acc[i], 0.0f);
}

// Segment sum / count / max by batch id (h >= 0 post-ReLU, so uint atomicMax
// on a zero-initialized buffer is an exact float max).
__global__ void pool_kernel(const float* __restrict__ h,
                            const int* __restrict__ batch,
                            float* __restrict__ sums,
                            float* __restrict__ mx,
                            float* __restrict__ cnt, int n)
{
    size_t tid = (size_t)blockIdx.x * blockDim.x + threadIdx.x;
    size_t i = tid >> 6;
    if (i >= (size_t)n) return;
    int f = (int)(tid & (FEAT - 1));
    int gidx = batch[i];
    float v = h[i * FEAT + f];
    atomicAdd(&sums[(size_t)gidx * FEAT + f], v);
    atomicMax((unsigned int*)&mx[(size_t)gidx * FEAT + f], __float_as_uint(v));
    if (f == 0) atomicAdd(&cnt[gidx], 1.0f);
}

__global__ void final_lin(const float* __restrict__ sums,
                          const float* __restrict__ mx,
                          const float* __restrict__ cnt,
                          const float* __restrict__ linW,
                          const float* __restrict__ linb,
                          float* __restrict__ out, int G)
{
    int tid = blockIdx.x * blockDim.x + threadIdx.x;
    if (tid >= G * OUTF) return;
    int g = tid / OUTF;
    int o = tid % OUTF;
    float inv = 1.0f / fmaxf(cnt[g], 1.0f);
    const float* wrow = linW + (size_t)o * (3 * FEAT);
    float acc = linb[o];
    for (int f = 0; f < FEAT; ++f) {
        float s = sums[(size_t)g * FEAT + f];
        acc += wrow[f] * s
             + wrow[FEAT + f] * (s * inv)
             + wrow[2 * FEAT + f] * mx[(size_t)g * FEAT + f];
    }
    out[tid] = acc;
}

// ---------------------------------------------------------------------------
static void run_cheb_layer(const float* in, const _Float16* in16,
                           const _Float16* Wt, const float* bias,
                           const int* row, const int* col, const float* wn,
                           float* sbuf, float* tx1, float* tx2, float* acc,
                           _Float16* a16,
                           int N, int NR, int E, float diagv, hipStream_t stream)
{
    const int gB = NR / 32;                       // unguarded WMMA grid
    const size_t NC = (size_t)N * FEAT;
    const int eB = (int)((NC + 255) / 256);
    const int sB = (int)(((long long)E * 16 + 255) / 256);
    const size_t ncb = NC * sizeof(float);

    // k = 0 : acc = in @ W0 + bias
    gemm_n64_f16wmma<<<gB, 256, 0, stream>>>(in16, Wt, bias, acc, 1);

    // k = 1 : Tx1 = L_hat @ in
    hipMemsetAsync(sbuf, 0, ncb, stream);
    scatter_lmul<<<sB, 256, 0, stream>>>(row, col, wn, in, sbuf, E);
    cheb_combine<<<eB, 256, 0, stream>>>(sbuf, in, in, tx1, a16, 1.0f, 0.0f, diagv, NC);
    gemm_n64_f16wmma<<<gB, 256, 0, stream>>>(a16, Wt + 4096, bias, acc, 0);

    // k = 2 : Tx2 = 2*L_hat@Tx1 - in
    hipMemsetAsync(sbuf, 0, ncb, stream);
    scatter_lmul<<<sB, 256, 0, stream>>>(row, col, wn, tx1, sbuf, E);
    cheb_combine<<<eB, 256, 0, stream>>>(sbuf, tx1, in, tx2, a16, 2.0f, -1.0f, diagv, NC);
    gemm_n64_f16wmma<<<gB, 256, 0, stream>>>(a16, Wt + 2 * 4096, bias, acc, 0);

    // k = 3 : Tx3 = 2*L_hat@Tx2 - Tx1  (overwrites Tx1)
    hipMemsetAsync(sbuf, 0, ncb, stream);
    scatter_lmul<<<sB, 256, 0, stream>>>(row, col, wn, tx2, sbuf, E);
    cheb_combine<<<eB, 256, 0, stream>>>(sbuf, tx2, tx1, tx1, a16, 2.0f, -1.0f, diagv, NC);
    gemm_n64_f16wmma<<<gB, 256, 0, stream>>>(a16, Wt + 3 * 4096, bias, acc, 0);
}

extern "C" void kernel_launch(void* const* d_in, const int* in_sizes, int n_in,
                              void* d_out, int out_size, void* d_ws, size_t ws_size,
                              hipStream_t stream)
{
    const float* x    = (const float*)d_in[0];
    const int*   ei   = (const int*)d_in[1];
    const float* ew   = (const float*)d_in[2];
    const int*   bat  = (const int*)d_in[3];
    const float* W1   = (const float*)d_in[4];
    const float* b1   = (const float*)d_in[5];
    const float* bn_g = (const float*)d_in[6];
    const float* bn_b = (const float*)d_in[7];
    const float* bn_m = (const float*)d_in[8];
    const float* bn_v = (const float*)d_in[9];
    const float* W2   = (const float*)d_in[10];
    const float* b2   = (const float*)d_in[11];
    const float* linW = (const float*)d_in[12];
    const float* linb = (const float*)d_in[13];
    float* out = (float*)d_out;

    const int N  = in_sizes[0] / FEAT;
    const int NR = (N + 31) & ~31;               // padded rows for guard-free WMMA
    const int E  = in_sizes[2];
    const int G  = out_size / OUTF;
    const int* row = ei;
    const int* col = ei + E;
    const float diagv = 2.0f / LMAX - 1.0f;

    // ---- carve workspace (256B aligned); node buffers padded to NR rows ----
    char* p = (char*)d_ws;
    auto carveB = [&](size_t nbytes) -> char* {
        char* r = p;
        p += ((nbytes + 255) & ~(size_t)255);
        return r;
    };
    float* deg  = (float*)carveB((size_t)N * sizeof(float));
    float* dis  = (float*)carveB((size_t)N * sizeof(float));
    float* wn   = (float*)carveB((size_t)E * sizeof(float));
    float* sbuf = (float*)carveB((size_t)NR * FEAT * sizeof(float));
    float* tx1  = (float*)carveB((size_t)NR * FEAT * sizeof(float));
    float* tx2  = (float*)carveB((size_t)NR * FEAT * sizeof(float));
    float* acc  = (float*)carveB((size_t)NR * FEAT * sizeof(float));
    float* hbuf = (float*)carveB((size_t)NR * FEAT * sizeof(float));
    _Float16* a16 = (_Float16*)carveB((size_t)NR * FEAT * sizeof(_Float16));
    _Float16* Wt1 = (_Float16*)carveB((size_t)4 * 4096 * sizeof(_Float16));
    _Float16* Wt2 = (_Float16*)carveB((size_t)4 * 4096 * sizeof(_Float16));
    float* psum = (float*)carveB((size_t)G * FEAT * sizeof(float));
    float* pmax = (float*)carveB((size_t)G * FEAT * sizeof(float));
    float* pcnt = (float*)carveB((size_t)G * sizeof(float));

    const size_t NC = (size_t)N * FEAT;
    const int eB = (int)((NC + 255) / 256);

    // ---- weight prep: f16, transposed so B fragments are contiguous ----
    wprep_kernel<<<(4 * 4096 + 255) / 256, 256, 0, stream>>>(W1, Wt1, 4 * 4096);
    wprep_kernel<<<(4 * 4096 + 255) / 256, 256, 0, stream>>>(W2, Wt2, 4 * 4096);

    // ---- symmetric normalization of L_hat ----
    hipMemsetAsync(deg, 0, N * sizeof(float), stream);
    deg_kernel<<<(E + 255) / 256, 256, 0, stream>>>(row, ew, deg, E);
    dis_kernel<<<(N + 255) / 256, 256, 0, stream>>>(deg, dis, N);
    wnorm_kernel<<<(E + 255) / 256, 256, 0, stream>>>(row, col, ew, dis, wn, E);

    // ---- layer 1: ChebConv -> ReLU -> BatchNorm (eval) ----
    cvt_f16_kernel<<<eB, 256, 0, stream>>>(x, a16, NC);
    run_cheb_layer(x, a16, Wt1, b1, row, col, wn, sbuf, tx1, tx2, acc, a16,
                   N, NR, E, diagv, stream);
    relu_bn_kernel<<<eB, 256, 0, stream>>>(acc, bn_g, bn_b, bn_m, bn_v,
                                           hbuf, a16, NC);

    // ---- layer 2: ChebConv -> ReLU (in place on acc) ----
    run_cheb_layer(hbuf, a16, Wt2, b2, row, col, wn, sbuf, tx1, tx2, acc, a16,
                   N, NR, E, diagv, stream);
    relu_inplace<<<eB, 256, 0, stream>>>(acc, NC);

    // ---- segment pooling (sum / mean / max) + final linear ----
    hipMemsetAsync(psum, 0, (size_t)G * FEAT * sizeof(float), stream);
    hipMemsetAsync(pmax, 0, (size_t)G * FEAT * sizeof(float), stream);
    hipMemsetAsync(pcnt, 0, G * sizeof(float), stream);
    pool_kernel<<<(int)((NC + 255) / 256), 256, 0, stream>>>(acc, bat, psum,
                                                             pmax, pcnt, N);
    final_lin<<<(G * OUTF + 255) / 256, 256, 0, stream>>>(psum, pmax, pcnt,
                                                          linW, linb, out, G);
}
